// ElmanRNN_60730837565949
// MI455X (gfx1250) — matrix operand order repaired and checked by reading, not compiled, not run on hardware
//
#include <hip/hip_runtime.h>
#include <math.h>

#define BATCH 4096
#define SEQL  512
#define INF   2
#define HID   128
#define HSTRIDE 136   // 272B row stride: b128 A-fragment loads hit all 64 LDS banks

typedef __attribute__((ext_vector_type(16))) __bf16 v16bf;
typedef __attribute__((ext_vector_type(8)))  __bf16 v8bf;
typedef __attribute__((ext_vector_type(8)))  float  v8f;

static __device__ __forceinline__ float fast_tanh(float v) {
#if __has_builtin(__builtin_amdgcn_tanhf)
    return __builtin_amdgcn_tanhf(v);     // v_tanh_f32 (TRANS, co-executes with WMMA)
#else
    return tanhf(v);
#endif
}

__global__ __launch_bounds__(32)
void ElmanRNN_wmma_kernel(const float* __restrict__ x,
                          const float* __restrict__ W_ih,
                          const float* __restrict__ b_ih,
                          const float* __restrict__ W_ho,
                          const float* __restrict__ b_ho,
                          float* __restrict__ out)
{
    __shared__ __bf16 hbuf[16 * HSTRIDE];     // wave-private h (16 rows x 128 feats, bf16)

    const int lane  = threadIdx.x;            // 0..31 (wave32)
    const int half  = lane >> 4;              // 0: lanes 0-15, 1: lanes 16-31
    const int nlane = lane & 15;
    const int row0  = blockIdx.x * 16;        // 16 batch rows per wave
    const float msk = (half == 0) ? 1.0f : 0.0f;

    // ---------------- Preload B fragments of W_h^T into VGPRs (loop-invariant) ---
    // B[k][n] = W_h[n][k] = W_ih[n][INF + k]; fragment lane n=nlane, k-half by `half`.
    v16bf Bf[8][4];
#pragma unroll
    for (int ti = 0; ti < 8; ++ti) {
        const float* wrow = W_ih + (size_t)(ti * 16 + nlane) * (INF + HID) + INF;
#pragma unroll
        for (int kk = 0; kk < 4; ++kk) {
            const float* wp = wrow + kk * 32 + half * 16;
            v16bf b;
#pragma unroll
            for (int e = 0; e < 16; ++e) b[e] = (__bf16)wp[e];
            Bf[ti][kk] = b;
        }
    }
    // Augmented K-chunk: rows {W_x[:,0], W_x[:,1], b_ih, 0...} (folds x-term + bias
    // into the matmul). Only lanes 0-15 (k=0..15 of the chunk) carry data.
    v16bf B5[8];
#pragma unroll
    for (int ti = 0; ti < 8; ++ti) {
        const int n = ti * 16 + nlane;
        v16bf b;
#pragma unroll
        for (int e = 0; e < 16; ++e) b[e] = (__bf16)0.0f;
        b[0] = (__bf16)(W_ih[(size_t)n * (INF + HID) + 0] * msk);
        b[1] = (__bf16)(W_ih[(size_t)n * (INF + HID) + 1] * msk);
        b[2] = (__bf16)(b_ih[n] * msk);
        B5[ti] = b;
    }

    // ---------------- h0 = 0 -------------------------------------------------------
    {
        unsigned int* hz = (unsigned int*)hbuf;
        for (int i = lane; i < (16 * HSTRIDE) / 2; i += 32) hz[i] = 0u;
    }

    // x prefetch (lane nlane owns batch row row0+nlane; halves load redundantly)
    const float* xrow = x + (size_t)(row0 + nlane) * (SEQL * INF);
    float2 xg = *(const float2*)(xrow);       // t = 0

    // ---------------- time loop: 512 sequential steps ------------------------------
    for (int t = 0; t < SEQL; ++t) {
        // Augmented A chunk: lanes 0-15 hold [x0, x1, 1, 0...]; lanes 16-31 all zero.
        v16bf a5;
#pragma unroll
        for (int e = 0; e < 16; ++e) a5[e] = (__bf16)0.0f;
        a5[0] = (__bf16)(xg.x * msk);
        a5[1] = (__bf16)(xg.y * msk);
        a5[2] = (__bf16)msk;

        // prefetch x for t+1 (overlaps the WMMAs below)
        const int tn = (t + 1 < SEQL) ? (t + 1) : t;
        xg = *(const float2*)(xrow + tn * INF);

        // A fragments of h (16x32 bf16 each), conflict-free ds_load_b128 pairs
        v16bf Af[4];
#pragma unroll
        for (int kk = 0; kk < 4; ++kk) {
            const __bf16* hp = hbuf + nlane * HSTRIDE + kk * 32 + half * 16;
            v8bf lo = *(const v8bf*)(hp);
            v8bf hi = *(const v8bf*)(hp + 8);
            Af[kk] = __builtin_shufflevector(lo, hi, 0,1,2,3,4,5,6,7,
                                                     8,9,10,11,12,13,14,15);
        }

        // 8 N-tiles x 5 WMMAs: h_new = tanh(h @ Wh^T + x @ Wx^T + b)
#pragma unroll
        for (int ti = 0; ti < 8; ++ti) {
            v8f c;
#pragma unroll
            for (int e = 0; e < 8; ++e) c[e] = 0.0f;
            c = __builtin_amdgcn_wmma_f32_16x16x32_bf16(false, a5, false, B5[ti],
                                                        (short)0, c, false, false);
#pragma unroll
            for (int kk = 0; kk < 4; ++kk)
                c = __builtin_amdgcn_wmma_f32_16x16x32_bf16(false, Af[kk], false,
                                                            Bf[ti][kk],
                                                            (short)0, c, false, false);
            // D layout: VGPR v -> (M = v + 8*half, N = ti*16 + nlane)
#pragma unroll
            for (int v = 0; v < 8; ++v) {
                const int m = v + half * 8;
                hbuf[m * HSTRIDE + ti * 16 + nlane] = (__bf16)fast_tanh(c[v]);
            }
        }
        // same-wave LDS ops are in-order: next iteration's Af loads see these stores
    }

    // ---------------- output head: out[b] = h_final @ W_ho^T + b_ho ----------------
    if (half == 0) {
        float acc = b_ho[0];
        const __bf16* hr = hbuf + nlane * HSTRIDE;
#pragma unroll 8
        for (int j = 0; j < HID; ++j) acc += (float)hr[j] * W_ho[j];
        out[row0 + nlane] = acc;
    }
}

extern "C" void kernel_launch(void* const* d_in, const int* in_sizes, int n_in,
                              void* d_out, int out_size, void* d_ws, size_t ws_size,
                              hipStream_t stream) {
    (void)in_sizes; (void)n_in; (void)d_ws; (void)ws_size; (void)out_size;
    const float* x    = (const float*)d_in[0];
    const float* W_ih = (const float*)d_in[1];
    const float* b_ih = (const float*)d_in[2];
    const float* W_ho = (const float*)d_in[3];
    const float* b_ho = (const float*)d_in[4];
    float* out = (float*)d_out;

    dim3 grid(BATCH / 16);   // 256 single-wave workgroups, 16 batch rows each
    dim3 block(32);          // one wave32
    ElmanRNN_wmma_kernel<<<grid, block, 0, stream>>>(x, W_ih, b_ih, W_ho, b_ho, out);
}
